// SINQLinear_88304527606015
// MI455X (gfx1250) — matrix operand order repaired and checked by reading, not compiled, not run on hardware
//
#include <hip/hip_runtime.h>

typedef _Float16 v16h __attribute__((ext_vector_type(16)));
typedef _Float16 v8h  __attribute__((ext_vector_type(8)));
typedef float    v8f  __attribute__((ext_vector_type(8)));
typedef int      async_v4i __attribute__((vector_size(16)));   // matches builtin param

#define M_DIM   8192
#define N_DIM   4096
#define K_DIM   4096
#define GROUP   64
#define NGROUPS (K_DIM / GROUP)

#define BM 256
#define BN 128
#define BK 32
#define KSTEPS (K_DIM / BK)
#define LDSW 40   // 32 halves + 8 pad -> 80B row stride, b128 conflict-free reads

#define AS1 __attribute__((address_space(1)))
#define AS3 __attribute__((address_space(3)))

#if defined(__has_builtin)
#if __has_builtin(__builtin_amdgcn_global_load_async_to_lds_b128) && \
    __has_builtin(__builtin_amdgcn_s_wait_asynccnt)
#define USE_ASYNC_LDS 1
#endif
#endif

typedef _Float16 lds_row_t[LDSW];

// ---------------------------------------------------------------------------
// Phase 1a: convert x (f32) -> f16, 8 elems/thread
// ---------------------------------------------------------------------------
__global__ __launch_bounds__(256)
void cvt_x_f16(const float* __restrict__ x, _Float16* __restrict__ xh) {
    const size_t i = ((size_t)blockIdx.x * 256 + threadIdx.x) * 8;
    const float4* p = (const float4*)(x + i);
    float4 a = p[0], b = p[1];
    v8h h;
    h[0] = (_Float16)a.x; h[1] = (_Float16)a.y; h[2] = (_Float16)a.z; h[3] = (_Float16)a.w;
    h[4] = (_Float16)b.x; h[5] = (_Float16)b.y; h[6] = (_Float16)b.z; h[7] = (_Float16)b.w;
    *(v8h*)(xh + i) = h;
}

// ---------------------------------------------------------------------------
// Phase 1b: dequantize W (packed 4-bit) -> f16, 8 packed bytes = 16 weights/thread
// ---------------------------------------------------------------------------
__global__ __launch_bounds__(256)
void dequant_w_f16(const int*   __restrict__ Wq,
                   const float* __restrict__ scale,
                   const float* __restrict__ zero,
                   const float* __restrict__ scale2,
                   _Float16*    __restrict__ wh) {
    const size_t t    = (size_t)blockIdx.x * 256 + threadIdx.x;
    const size_t base = t * 8;                       // int (byte) index
    const int n   = (int)(base >> 11);               // K/2 = 2048 ints per row
    const int kbc = (int)(base & 2047);
    const int k0  = kbc * 2;
    const int g   = k0 >> 6;                         // 16 k-values lie in one group
    const float sc = scale[(size_t)n * NGROUPS + g] * scale2[n];
    const float zp = zero[(size_t)n * NGROUPS + g];

    const int4* pw = (const int4*)(Wq + base);
    int4 wa0 = pw[0], wa1 = pw[1];
    int vals[8] = { wa0.x, wa0.y, wa0.z, wa0.w, wa1.x, wa1.y, wa1.z, wa1.w };

    v8h b0, b1;
#pragma unroll
    for (int i = 0; i < 8; ++i) {
        const int v = vals[i];
        const _Float16 wlo = (_Float16)(((float)(v & 0xF)        - zp) * sc);
        const _Float16 whi = (_Float16)(((float)((v >> 4) & 0xF) - zp) * sc);
        if (i < 4) { b0[2 * i]       = wlo; b0[2 * i + 1]       = whi; }
        else       { b1[2 * (i - 4)] = wlo; b1[2 * (i - 4) + 1] = whi; }
    }
    _Float16* dst = wh + (size_t)n * K_DIM + k0;
    *(v8h*)dst       = b0;
    *(v8h*)(dst + 8) = b1;
}

// ---------------------------------------------------------------------------
// Phase 2: pure f16 GEMM with copy-only staging (async global->LDS if available)
// ---------------------------------------------------------------------------
__global__ __launch_bounds__(256)
void gemm_f16_wmma(const _Float16* __restrict__ xh,
                   const _Float16* __restrict__ wh,
                   float*          __restrict__ out) {
    __shared__ _Float16 As[2][BM][LDSW];   // 40 KB
    __shared__ _Float16 Bs[2][BN][LDSW];   // 20 KB

    const int tid  = threadIdx.x;
    const int lane = tid & 31;
    const int wave = tid >> 5;
    const int wm   = wave >> 1;            // 0..3 : 64-row band
    const int wn   = wave & 1;             // 0..1 : 64-col band

    const int block_m = blockIdx.y * BM;
    const int block_n = blockIdx.x * BN;

    // staging: A row = tid (32 halves); B row = tid>>1, 16-half half = tid&1
    const _Float16* gxrow = xh + (size_t)(block_m + tid) * K_DIM;
    const int brow  = tid >> 1;
    const int bhalf = tid & 1;
    const _Float16* gwrow = wh + (size_t)(block_n + brow) * K_DIM + bhalf * 16;

#if defined(USE_ASYNC_LDS)
    auto stage = [&](lds_row_t* Ad, lds_row_t* Bd, int kt) {
        const _Float16* gx = gxrow + (size_t)kt * BK;
        _Float16* lx = &Ad[tid][0];
#pragma unroll
        for (int c = 0; c < 4; ++c)
            __builtin_amdgcn_global_load_async_to_lds_b128(
                (AS1 async_v4i*)(void*)(gx + c * 8),
                (AS3 async_v4i*)(void*)(lx + c * 8), 0, 0);
        const _Float16* gw = gwrow + (size_t)kt * BK;
        _Float16* lw = &Bd[brow][bhalf * 16];
#pragma unroll
        for (int c = 0; c < 2; ++c)
            __builtin_amdgcn_global_load_async_to_lds_b128(
                (AS1 async_v4i*)(void*)(gw + c * 8),
                (AS3 async_v4i*)(void*)(lw + c * 8), 0, 0);
    };
#else
    float4 ra[4], rb[2];                   // raw f16 payload (copy-only)
    auto stage_load = [&](int kt) {
        const float4* gx = (const float4*)(gxrow + (size_t)kt * BK);
#pragma unroll
        for (int c = 0; c < 4; ++c) ra[c] = gx[c];
        const float4* gw = (const float4*)(gwrow + (size_t)kt * BK);
        rb[0] = gw[0]; rb[1] = gw[1];
    };
    auto stage_store = [&](lds_row_t* Ad, lds_row_t* Bd) {
        float4* lx = (float4*)&Ad[tid][0];
#pragma unroll
        for (int c = 0; c < 4; ++c) lx[c] = ra[c];
        float4* lw = (float4*)&Bd[brow][bhalf * 16];
        lw[0] = rb[0]; lw[1] = rb[1];
    };
#endif

    // WMMA fragment loads (ISA §7.12.2 layouts)
    auto frag_a = [&](const lds_row_t* A, int mi) -> v16h {
        const int r  = wm * 64 + mi * 16 + (lane & 15);
        const int kb = (lane >> 4) * 8;
        const v8h lo = *(const v8h*)&A[r][kb];
        const v8h hi = *(const v8h*)&A[r][kb + 16];
        return __builtin_shufflevector(lo, hi, 0,1,2,3,4,5,6,7,8,9,10,11,12,13,14,15);
    };
    auto frag_b = [&](const lds_row_t* B, int ni) -> v16h {
        const int c  = wn * 64 + ni * 16 + (lane & 15);
        const int kb = (lane >> 4) * 8;
        const v8h lo = *(const v8h*)&B[c][kb];
        const v8h hi = *(const v8h*)&B[c][kb + 16];
        return __builtin_shufflevector(lo, hi, 0,1,2,3,4,5,6,7,8,9,10,11,12,13,14,15);
    };

    v8f acc[4][4] = {};

    lds_row_t* Acur = As[0]; lds_row_t* Anxt = As[1];
    lds_row_t* Bcur = Bs[0]; lds_row_t* Bnxt = Bs[1];

    // prologue: stage buffer 0
#if defined(USE_ASYNC_LDS)
    stage(Acur, Bcur, 0);
    __builtin_amdgcn_s_wait_asynccnt(0);
#else
    stage_load(0);
    stage_store(Acur, Bcur);
#endif
    __syncthreads();

#pragma unroll 1
    for (int kt = 0; kt < KSTEPS; ++kt) {
        const bool has_next = (kt + 1) < KSTEPS;

        if (has_next) {
#if defined(USE_ASYNC_LDS)
            stage(Anxt, Bnxt, kt + 1);             // async writes to other buffer
#else
            stage_load(kt + 1);
            __builtin_prefetch(gxrow + (size_t)(kt + 2) * BK, 0, 0);
            __builtin_prefetch(gwrow + (size_t)(kt + 2) * BK, 0, 0);
#endif
        }

        v16h b0 = frag_b(Bcur, 0);
        v16h b1 = frag_b(Bcur, 1);
        v16h b2 = frag_b(Bcur, 2);
        v16h b3 = frag_b(Bcur, 3);

        // software-pipelined A fragments: load mi+1 before mi's WMMA group
        v16h a_cur = frag_a(Acur, 0);
#pragma unroll
        for (int mi = 0; mi < 4; ++mi) {
            v16h a_nxt = (mi < 3) ? frag_a(Acur, mi + 1) : a_cur;
            acc[mi][0] = __builtin_amdgcn_wmma_f32_16x16x32_f16(false, a_cur, false, b0, (short)0, acc[mi][0], false, false);
            acc[mi][1] = __builtin_amdgcn_wmma_f32_16x16x32_f16(false, a_cur, false, b1, (short)0, acc[mi][1], false, false);
            acc[mi][2] = __builtin_amdgcn_wmma_f32_16x16x32_f16(false, a_cur, false, b2, (short)0, acc[mi][2], false, false);
            acc[mi][3] = __builtin_amdgcn_wmma_f32_16x16x32_f16(false, a_cur, false, b3, (short)0, acc[mi][3], false, false);
            a_cur = a_nxt;
        }

#if defined(USE_ASYNC_LDS)
        if (has_next) __builtin_amdgcn_s_wait_asynccnt(0);   // my LDS writes landed
#else
        if (has_next) stage_store(Anxt, Bnxt);
#endif
        __syncthreads();   // single barrier per K-step

        lds_row_t* t0 = Acur; Acur = Anxt; Anxt = t0;
        lds_row_t* t1 = Bcur; Bcur = Bnxt; Bnxt = t1;
    }

    // writeback: f32 C/D layout: VGPR r -> M = r + (lane>=16 ? 8 : 0), N = lane&15
    const int coln = (lane & 15);
    const int rowh = (lane >> 4) * 8;
#pragma unroll
    for (int mi = 0; mi < 4; ++mi) {
#pragma unroll
        for (int ni = 0; ni < 4; ++ni) {
            const int row0 = block_m + wm * 64 + mi * 16 + rowh;
            const int col  = block_n + wn * 64 + ni * 16 + coln;
            float* po = out + (size_t)row0 * N_DIM + col;
#pragma unroll
            for (int r = 0; r < 8; ++r) po[(size_t)r * N_DIM] = acc[mi][ni][r];
        }
    }
}

// ---------------------------------------------------------------------------
// Fallback: fused dequant + GEMM if workspace is too small
// ---------------------------------------------------------------------------
__global__ __launch_bounds__(256)
void sinq_w4_fused(const float* __restrict__ x,
                   const int*   __restrict__ Wq,
                   const float* __restrict__ scale,
                   const float* __restrict__ zero,
                   const float* __restrict__ scale2,
                   float*       __restrict__ out) {
    __shared__ _Float16 As[2][BM][LDSW];
    __shared__ _Float16 Bs[2][BN][LDSW];

    const int tid  = threadIdx.x;
    const int lane = tid & 31;
    const int wave = tid >> 5;
    const int wm   = wave >> 1;
    const int wn   = wave & 1;
    const int block_m = blockIdx.y * BM;
    const int block_n = blockIdx.x * BN;

    const float* xrow = x + (size_t)(block_m + tid) * K_DIM;
    const int brow  = tid >> 1;
    const int bhalf = tid & 1;
    const int*   wrow = Wq + (size_t)(block_n + brow) * (K_DIM / 2) + bhalf * 8;
    const float* brow_scale = scale + (size_t)(block_n + brow) * NGROUPS;
    const float* brow_zero  = zero  + (size_t)(block_n + brow) * NGROUPS;
    const float  s2 = scale2[block_n + brow];

    float4 xa[8];
    int4   wa[2];

    auto load_global = [&](int kt) {
        const float4* px = (const float4*)(xrow + (size_t)kt * BK);
#pragma unroll
        for (int i = 0; i < 8; ++i) xa[i] = px[i];
        const int4* pw = (const int4*)(wrow + (size_t)kt * (BK / 2));
        wa[0] = pw[0]; wa[1] = pw[1];
    };
    auto store_lds = [&](lds_row_t* Ad, lds_row_t* Bd, int kt) {
        const float* xf = (const float*)xa;
        _Float16* ap = &Ad[tid][0];
#pragma unroll
        for (int v = 0; v < 4; ++v) {
            v8h h;
#pragma unroll
            for (int i = 0; i < 8; ++i) h[i] = (_Float16)xf[v * 8 + i];
            *(v8h*)(ap + v * 8) = h;
        }
        const int g  = (kt * BK) / GROUP;
        const float sc = brow_scale[g] * s2;
        const float zp = brow_zero[g];
        const int* wv = (const int*)wa;
        v8h b0, b1;
#pragma unroll
        for (int i = 0; i < 8; ++i) {
            const int v = wv[i];
            const _Float16 wlo = (_Float16)(((float)(v & 0xF)        - zp) * sc);
            const _Float16 whi = (_Float16)(((float)((v >> 4) & 0xF) - zp) * sc);
            if (i < 4) { b0[2 * i]       = wlo; b0[2 * i + 1]       = whi; }
            else       { b1[2 * (i - 4)] = wlo; b1[2 * (i - 4) + 1] = whi; }
        }
        _Float16* bp = &Bd[brow][bhalf * 16];
        *(v8h*)bp       = b0;
        *(v8h*)(bp + 8) = b1;
    };
    auto frag_a = [&](const lds_row_t* A, int mi) -> v16h {
        const int r  = wm * 64 + mi * 16 + (lane & 15);
        const int kb = (lane >> 4) * 8;
        const v8h lo = *(const v8h*)&A[r][kb];
        const v8h hi = *(const v8h*)&A[r][kb + 16];
        return __builtin_shufflevector(lo, hi, 0,1,2,3,4,5,6,7,8,9,10,11,12,13,14,15);
    };
    auto frag_b = [&](const lds_row_t* B, int ni) -> v16h {
        const int c  = wn * 64 + ni * 16 + (lane & 15);
        const int kb = (lane >> 4) * 8;
        const v8h lo = *(const v8h*)&B[c][kb];
        const v8h hi = *(const v8h*)&B[c][kb + 16];
        return __builtin_shufflevector(lo, hi, 0,1,2,3,4,5,6,7,8,9,10,11,12,13,14,15);
    };

    v8f acc[4][4] = {};
    lds_row_t* Acur = As[0]; lds_row_t* Anxt = As[1];
    lds_row_t* Bcur = Bs[0]; lds_row_t* Bnxt = Bs[1];

    load_global(0);
    store_lds(Acur, Bcur, 0);
    __syncthreads();

#pragma unroll 1
    for (int kt = 0; kt < KSTEPS; ++kt) {
        if (kt + 1 < KSTEPS) load_global(kt + 1);

        v16h b0 = frag_b(Bcur, 0);
        v16h b1 = frag_b(Bcur, 1);
        v16h b2 = frag_b(Bcur, 2);
        v16h b3 = frag_b(Bcur, 3);

        v16h a_cur = frag_a(Acur, 0);
#pragma unroll
        for (int mi = 0; mi < 4; ++mi) {
            v16h a_nxt = (mi < 3) ? frag_a(Acur, mi + 1) : a_cur;
            acc[mi][0] = __builtin_amdgcn_wmma_f32_16x16x32_f16(false, a_cur, false, b0, (short)0, acc[mi][0], false, false);
            acc[mi][1] = __builtin_amdgcn_wmma_f32_16x16x32_f16(false, a_cur, false, b1, (short)0, acc[mi][1], false, false);
            acc[mi][2] = __builtin_amdgcn_wmma_f32_16x16x32_f16(false, a_cur, false, b2, (short)0, acc[mi][2], false, false);
            acc[mi][3] = __builtin_amdgcn_wmma_f32_16x16x32_f16(false, a_cur, false, b3, (short)0, acc[mi][3], false, false);
            a_cur = a_nxt;
        }

        if (kt + 1 < KSTEPS) store_lds(Anxt, Bnxt, kt + 1);
        __syncthreads();

        lds_row_t* t0 = Acur; Acur = Anxt; Anxt = t0;
        lds_row_t* t1 = Bcur; Bcur = Bnxt; Bnxt = t1;
    }

    const int coln = (lane & 15);
    const int rowh = (lane >> 4) * 8;
#pragma unroll
    for (int mi = 0; mi < 4; ++mi) {
#pragma unroll
        for (int ni = 0; ni < 4; ++ni) {
            const int row0 = block_m + wm * 64 + mi * 16 + rowh;
            const int col  = block_n + wn * 64 + ni * 16 + coln;
            float* po = out + (size_t)row0 * N_DIM + col;
#pragma unroll
            for (int r = 0; r < 8; ++r) po[(size_t)r * N_DIM] = acc[mi][ni][r];
        }
    }
}

// ---------------------------------------------------------------------------
extern "C" void kernel_launch(void* const* d_in, const int* in_sizes, int n_in,
                              void* d_out, int out_size, void* d_ws, size_t ws_size,
                              hipStream_t stream) {
    (void)in_sizes; (void)n_in; (void)out_size;
    const float* x      = (const float*)d_in[0];
    const int*   Wq     = (const int*)d_in[1];
    const float* scale  = (const float*)d_in[2];
    const float* zero   = (const float*)d_in[3];
    const float* scale2 = (const float*)d_in[4];
    float* out = (float*)d_out;

    const size_t xh_bytes = (size_t)M_DIM * K_DIM * sizeof(_Float16);   // 64 MB
    const size_t wh_bytes = (size_t)N_DIM * K_DIM * sizeof(_Float16);   // 32 MB

    if (ws_size >= xh_bytes + wh_bytes) {
        _Float16* xh = (_Float16*)d_ws;
        _Float16* wh = (_Float16*)((char*)d_ws + xh_bytes);
        cvt_x_f16<<<dim3((M_DIM * (size_t)K_DIM) / 8 / 256), dim3(256), 0, stream>>>(x, xh);
        dequant_w_f16<<<dim3((N_DIM * (size_t)K_DIM / 2) / 8 / 256), dim3(256), 0, stream>>>(
            Wq, scale, zero, scale2, wh);
        gemm_f16_wmma<<<dim3(N_DIM / BN, M_DIM / BM), dim3(256), 0, stream>>>(xh, wh, out);
    } else {
        sinq_w4_fused<<<dim3(N_DIM / BN, M_DIM / BM), dim3(256), 0, stream>>>(
            x, Wq, scale, zero, scale2, out);
    }
}
